// FlashAttention_16114717295069
// MI455X (gfx1250) — matrix operand order repaired and checked by reading, not compiled, not run on hardware
//
#include <hip/hip_runtime.h>
#include <hip/hip_bf16.h>

typedef __attribute__((ext_vector_type(16))) __bf16 v16bf;
typedef __attribute__((ext_vector_type(8)))  __bf16 v8bf;
typedef __attribute__((ext_vector_type(8)))  float  v8f;

#define DIM_C   2048
#define HEADS_C 16
#define HD_C    128
#define MROWS_C 4096   // B*S
#define SEQ_C   2048

// ---------------- elementwise f32 -> bf16 ----------------
__global__ void cvt_bf16_kernel(const float* __restrict__ in, __bf16* __restrict__ out, int n) {
    int i = blockIdx.x * blockDim.x + threadIdx.x;
    if (i < n) out[i] = (__bf16)in[i];
}

// ---------------- transpose + convert: W[K][N] f32 -> WT[N][K] bf16 ----------------
__global__ void transpose_cvt_kernel(const float* __restrict__ W, __bf16* __restrict__ WT,
                                     int K, int N) {
    int i = blockIdx.x * blockDim.x + threadIdx.x;
    int total = K * N;
    if (i < total) {
        int n = i / K;
        int k = i - n * K;
        WT[i] = (__bf16)W[(size_t)k * N + n];
    }
}

// ---------------- WMMA GEMM: C[M][N] = A[M][K] * BT[N][K]^T + bias ----------------
// Per wave: 32 (M) x 64 (N) output tile -> 8 WMMAs per 32-wide k-step vs 12 b128 loads.
// Block = 256 threads = 8 waves covering 256 M rows.
__global__ __launch_bounds__(256) void gemm_wmma_kernel(
    const __bf16* __restrict__ A, const __bf16* __restrict__ BT,
    const float* __restrict__ bias,
    float* __restrict__ outF,         // f32 row-major (or null)
    __bf16* __restrict__ outB,        // bf16 row-major, scaled (or null)
    __bf16* __restrict__ outT,        // bf16 transposed [N][M] (or null)
    int M, int N, int K, float scale)
{
    const int lane = threadIdx.x & 31;
    const int wave = threadIdx.x >> 5;
    const int g    = lane >> 4;       // half-wave group
    const int ln   = lane & 15;
    const int m0   = blockIdx.y * 256 + wave * 32;
    const int n0   = blockIdx.x * 64;

    v8f acc[2][4] = {};
    const __bf16* arow0 = A + (size_t)(m0 + ln) * K;
    const __bf16* arow1 = A + (size_t)(m0 + 16 + ln) * K;

    for (int k = 0; k < K; k += 32) {
        // A fragments (16x32 each): elems 0..15 <- K = k+16g .. k+16g+15, contiguous
        v16bf af0, af1;
        {
            v8bf lo0 = *(const v8bf*)(arow0 + k + 16 * g);
            v8bf hi0 = *(const v8bf*)(arow0 + k + 16 * g + 8);
            v8bf lo1 = *(const v8bf*)(arow1 + k + 16 * g);
            v8bf hi1 = *(const v8bf*)(arow1 + k + 16 * g + 8);
            #pragma unroll
            for (int i = 0; i < 8; ++i) {
                af0[i] = lo0[i]; af0[i + 8] = hi0[i];
                af1[i] = lo1[i]; af1[i + 8] = hi1[i];
            }
        }

        #pragma unroll
        for (int ns = 0; ns < 4; ++ns) {
            // B fragment (32x16): lane holds BT[n][k+16g .. +15], contiguous
            v16bf bfr = *(const v16bf*)(BT + (size_t)(n0 + ns * 16 + ln) * K + k + 16 * g);
            acc[0][ns] = __builtin_amdgcn_wmma_f32_16x16x32_bf16(
                false, af0, false, bfr, (short)0, acc[0][ns], false, false);
            acc[1][ns] = __builtin_amdgcn_wmma_f32_16x16x32_bf16(
                false, af1, false, bfr, (short)0, acc[1][ns], false, false);
        }
    }

    #pragma unroll
    for (int ms = 0; ms < 2; ++ms) {
        #pragma unroll
        for (int ns = 0; ns < 4; ++ns) {
            int n = n0 + ns * 16 + ln;
            float bv = bias ? bias[n] : 0.0f;
            #pragma unroll
            for (int r = 0; r < 8; ++r) {
                int m = m0 + ms * 16 + r + 8 * g;  // C layout: VGPR r -> M = r + 8g, N = lane&15
                float v = acc[ms][ns][r] + bv;
                if (outF) outF[(size_t)m * N + n] = v;
                if (outB) outB[(size_t)m * N + n] = (__bf16)(v * scale);
                if (outT) outT[(size_t)n * M + m] = (__bf16)v;
            }
        }
    }
}

// ---------------- Flash MQA attention ----------------
// Grid: one block per 16-query tile (B*S/16 = 256). Block = 512 threads = 16 waves = 16 heads.
// Qb: bf16 [4096][2048] (pre-scaled by 1/sqrt(D)); Kb: bf16 [4096][128]; Vt: bf16 [128][4096].
// Ob: bf16 [4096][2048].
__global__ __launch_bounds__(512) void flash_mqa_kernel(
    const __bf16* __restrict__ Qb, const __bf16* __restrict__ Kb,
    const __bf16* __restrict__ Vt, __bf16* __restrict__ Ob)
{
    __shared__ __bf16 pbuf[16 * 16 * 32];   // 16 waves x (16x32) P tile

    const int lane = threadIdx.x & 31;
    const int h    = threadIdx.x >> 5;      // head
    const int g    = lane >> 4;
    const int ln   = lane & 15;
    const int t    = blockIdx.x;            // query tile: rows t*16 .. t*16+15 (global m)
    const int b    = t >> 7;                // batch (t*16 / 2048)
    const size_t mbase = (size_t)t * 16;

    // Q fragments: 16x128 as four 16x32 A-fragments
    v16bf qa[4];
    {
        const __bf16* qrow = Qb + (mbase + ln) * DIM_C + h * HD_C;
        #pragma unroll
        for (int kk = 0; kk < 4; ++kk) {
            v8bf lo = *(const v8bf*)(qrow + kk * 32 + 16 * g);
            v8bf hi = *(const v8bf*)(qrow + kk * 32 + 16 * g + 8);
            #pragma unroll
            for (int i = 0; i < 8; ++i) { qa[kk][i] = lo[i]; qa[kk][i + 8] = hi[i]; }
        }
    }

    v8f o_acc[8] = {};
    float row_max[8], row_sum[8];
    #pragma unroll
    for (int r = 0; r < 8; ++r) { row_max[r] = -3.0e38f; row_sum[r] = 0.0f; }

    __bf16* pw = pbuf + h * 512;                     // this wave's 16x32 P tile
    const __bf16* kbb = Kb + (size_t)b * SEQ_C * HD_C;
    const __bf16* vtb = Vt + (size_t)b * SEQ_C;      // vt[d][m]: stride MROWS_C per d

    for (int kb = 0; kb < SEQ_C; kb += 32) {
        // --- scores: two 16x16 tiles (keys kb..+15, kb+16..+31) ---
        v8f c0 = {}, c1 = {};
        #pragma unroll
        for (int kk = 0; kk < 4; ++kk) {
            v16bf b0 = *(const v16bf*)(kbb + (size_t)(kb + ln) * HD_C + kk * 32 + 16 * g);
            c0 = __builtin_amdgcn_wmma_f32_16x16x32_bf16(
                false, qa[kk], false, b0, (short)0, c0, false, false);
            v16bf b1 = *(const v16bf*)(kbb + (size_t)(kb + 16 + ln) * HD_C + kk * 32 + 16 * g);
            c1 = __builtin_amdgcn_wmma_f32_16x16x32_bf16(
                false, qa[kk], false, b1, (short)0, c1, false, false);
        }

        // --- online softmax per row (rows live across 16-lane groups) ---
        #pragma unroll
        for (int r = 0; r < 8; ++r) {
            float s0 = c0[r], s1 = c1[r];
            float mx = fmaxf(s0, s1);
            #pragma unroll
            for (int o = 1; o < 16; o <<= 1) mx = fmaxf(mx, __shfl_xor(mx, o));
            float mnew = fmaxf(row_max[r], mx);
            float corr = __expf(row_max[r] - mnew);
            float p0 = __expf(s0 - mnew);
            float p1 = __expf(s1 - mnew);
            float ps = p0 + p1;
            #pragma unroll
            for (int o = 1; o < 16; o <<= 1) ps += __shfl_xor(ps, o);
            row_sum[r] = row_sum[r] * corr + ps;
            row_max[r] = mnew;
            #pragma unroll
            for (int dt = 0; dt < 8; ++dt) o_acc[dt][r] *= corr;
            int row = r + 8 * g;                 // C layout row
            pw[row * 32 + ln]      = (__bf16)p0; // keys 0..15 of tile
            pw[row * 32 + 16 + ln] = (__bf16)p1; // keys 16..31
        }

        // --- read P back in A-fragment layout (same wave; LDS in-order) ---
        v16bf pf;
        {
            const __bf16* pr = pw + ln * 32;
            v8bf lo = *(const v8bf*)(pr + 16 * g);
            v8bf hi = *(const v8bf*)(pr + 16 * g + 8);
            #pragma unroll
            for (int i = 0; i < 8; ++i) { pf[i] = lo[i]; pf[i + 8] = hi[i]; }
        }

        // --- O += P(16x32) x V(32x128): 8 WMMAs over d-tiles ---
        #pragma unroll
        for (int dt = 0; dt < 8; ++dt) {
            v16bf vf = *(const v16bf*)(vtb + (size_t)(dt * 16 + ln) * MROWS_C + kb + 16 * g);
            o_acc[dt] = __builtin_amdgcn_wmma_f32_16x16x32_bf16(
                false, pf, false, vf, (short)0, o_acc[dt], false, false);
        }
    }

    // --- normalize and store bf16 row-major [4096][2048] ---
    #pragma unroll
    for (int dt = 0; dt < 8; ++dt) {
        #pragma unroll
        for (int r = 0; r < 8; ++r) {
            int row = r + 8 * g;
            float v = o_acc[dt][r] / row_sum[r];
            Ob[(mbase + row) * DIM_C + h * HD_C + dt * 16 + ln] = (__bf16)v;
        }
    }
}

extern "C" void kernel_launch(void* const* d_in, const int* in_sizes, int n_in,
                              void* d_out, int out_size, void* d_ws, size_t ws_size,
                              hipStream_t stream) {
    const float* x  = (const float*)d_in[0];
    const float* Wq = (const float*)d_in[1];
    const float* bq = (const float*)d_in[2];
    const float* Wk = (const float*)d_in[3];
    const float* bk = (const float*)d_in[4];
    const float* Wv = (const float*)d_in[5];
    const float* bv = (const float*)d_in[6];
    const float* Wo = (const float*)d_in[7];
    const float* bo = (const float*)d_in[8];
    float* out = (float*)d_out;

    const int M = MROWS_C, Kd = DIM_C, Nq = DIM_C, Dh = HD_C;

    char* ws = (char*)d_ws;
    size_t off = 0;
    auto carve = [&](size_t bytes) -> char* {
        char* p = ws + off;
        off += (bytes + 255) & ~(size_t)255;
        return p;
    };
    __bf16* xb  = (__bf16*)carve((size_t)M * Kd * 2);
    __bf16* wqT = (__bf16*)carve((size_t)Nq * Kd * 2);
    __bf16* wkT = (__bf16*)carve((size_t)Dh * Kd * 2);
    __bf16* wvT = (__bf16*)carve((size_t)Dh * Kd * 2);
    __bf16* woT = (__bf16*)carve((size_t)Kd * Nq * 2);
    __bf16* qb  = (__bf16*)carve((size_t)M * Nq * 2);
    __bf16* kbf = (__bf16*)carve((size_t)M * Dh * 2);
    __bf16* vt  = (__bf16*)carve((size_t)Dh * M * 2);
    __bf16* ab  = (__bf16*)carve((size_t)M * Nq * 2);
    (void)ws_size; (void)in_sizes; (void)n_in; (void)out_size;

    // Stage 0: precision conversion / weight transposition (stays resident in L2)
    cvt_bf16_kernel<<<(M * Kd) / 256, 256, 0, stream>>>(x, xb, M * Kd);
    transpose_cvt_kernel<<<(Kd * Nq) / 256, 256, 0, stream>>>(Wq, wqT, Kd, Nq);
    transpose_cvt_kernel<<<(Kd * Dh) / 256, 256, 0, stream>>>(Wk, wkT, Kd, Dh);
    transpose_cvt_kernel<<<(Kd * Dh) / 256, 256, 0, stream>>>(Wv, wvT, Kd, Dh);
    transpose_cvt_kernel<<<(Nq * Kd) / 256, 256, 0, stream>>>(Wo, woT, Nq, Kd);

    const float inv_sqrt_d = 0.08838834764831845f;  // 1/sqrt(128)

    // Stage 1: projections (WMMA bf16)
    gemm_wmma_kernel<<<dim3(Nq / 64, M / 256), 256, 0, stream>>>(
        xb, wqT, bq, nullptr, qb, nullptr, M, Nq, Kd, inv_sqrt_d);   // Q, pre-scaled
    gemm_wmma_kernel<<<dim3(Dh / 64, M / 256), 256, 0, stream>>>(
        xb, wkT, bk, nullptr, kbf, nullptr, M, Dh, Kd, 1.0f);        // K row-major
    gemm_wmma_kernel<<<dim3(Dh / 64, M / 256), 256, 0, stream>>>(
        xb, wvT, bv, nullptr, nullptr, vt, M, Dh, Kd, 1.0f);         // V transposed [D][M]

    // Stage 2: flash attention (16 heads per block, shared K/V streams)
    flash_mqa_kernel<<<M / 16, 512, 0, stream>>>(qb, kbf, vt, ab);

    // Stage 3: output projection (WMMA bf16, f32 out + bias)
    gemm_wmma_kernel<<<dim3(Kd / 64, M / 256), 256, 0, stream>>>(
        ab, woT, bo, out, nullptr, nullptr, M, Kd, Nq, 1.0f);
}
